// SelfAttentionLayer_77025943486575
// MI455X (gfx1250) — matrix operand order repaired and checked by reading, not compiled, not run on hardware
//
#include <hip/hip_runtime.h>
#include <hip/hip_bf16.h>
#include <math.h>

typedef __attribute__((ext_vector_type(16))) _Float16 v16h;
typedef __attribute__((ext_vector_type(8)))  _Float16 v8h;
typedef __attribute__((ext_vector_type(8)))  float    v8f;
typedef __attribute__((ext_vector_type(4)))  float    v4f;
typedef __attribute__((ext_vector_type(4)))  int      v4i;

#define BS_ROWS 8192   // B*S
#define D_DIM   512
#define U_DIM   256
#define S_LEN   2048

// --------------------------------------------------------------------------
// CDNA5 async global->LDS staging (guarded; falls back to sync vector copy).
// Builtin prototype (probe-confirmed): (int4 AS1*, int4 AS3*, Imm, Imm).
// --------------------------------------------------------------------------
#if defined(__gfx1250__) && __has_builtin(__builtin_amdgcn_global_load_async_to_lds_b128) && __has_builtin(__builtin_amdgcn_s_wait_asynccnt)
#define USE_ASYNC_LDS 1
typedef __attribute__((address_space(1))) v4i* as1_v4i_p;
typedef __attribute__((address_space(3))) v4i* as3_v4i_p;
__device__ __forceinline__ void async_copy_b128(const void* g, void* l) {
    // generic->AS1 is numerically identity; generic LDS ptr low 32 bits are the
    // AS3 offset (aperture tag lives in the high bits).
    __builtin_amdgcn_global_load_async_to_lds_b128(
        (as1_v4i_p)(unsigned long long)g,
        (as3_v4i_p)(unsigned int)(unsigned long long)l,
        0, 0);
}
#endif

__device__ __forceinline__ v16h loadB16(const _Float16* p) {
    v8h lo = *(const v8h*)(p);
    v8h hi = *(const v8h*)(p + 8);
    v16h b;
#pragma unroll
    for (int e = 0; e < 8; ++e) { b[e] = lo[e]; b[8 + e] = hi[e]; }
    return b;
}

// --------------------------------------------------------------------------
// Kernel 0: one-time W transpose+convert:  WT[z][u][d] = (f16) W_z[d][u]
// --------------------------------------------------------------------------
__global__ __launch_bounds__(256)
void wt_kernel(const float* __restrict__ Wq, const float* __restrict__ Wk,
               const float* __restrict__ Wv, _Float16* __restrict__ WT) {
    const int idx = blockIdx.x * 256 + threadIdx.x;   // over 3*256*512
    const int d = idx & (D_DIM - 1);
    const int u = (idx >> 9) & (U_DIM - 1);
    const int z = idx >> 17;
    const float* W = (z == 0) ? Wq : ((z == 1) ? Wk : Wv);
    WT[idx] = (_Float16)W[(size_t)d * U_DIM + u];
}

// --------------------------------------------------------------------------
// Kernel 1: projections.  One wave per (16-row x strip, matrix z); the strip
// is held resident as 16 A-fragments and reused for all 16 column tiles.
// B-fragment loads are software-pipelined one chunk ahead, and accumulation
// uses two independent WMMA chains to cover XDL latency.
// --------------------------------------------------------------------------
__global__ __launch_bounds__(128)
void proj_qkv_kernel(const float* __restrict__ x,
                     const _Float16* __restrict__ WT,
                     _Float16* __restrict__ qd,
                     _Float16* __restrict__ kd,
                     _Float16* __restrict__ vd,
                     _Float16* __restrict__ vT) {
    const int lane = threadIdx.x & 31;
    const int wave = threadIdx.x >> 5;
    const int rowTile = blockIdx.x * 4 + wave;        // 0..511
    const int z = blockIdx.y;                         // 0..2
    const int m = lane & 15;
    const int h = lane >> 4;
    const int R0 = rowTile * 16;

    _Float16* dst = (z == 0) ? qd : ((z == 1) ? kd : vd);
    const _Float16* WTz = WT + (size_t)z * U_DIM * D_DIM;

    // Preload x strip as 16 A-fragments (K chunks of 32 over D=512).
    v16h xa[16];
    const float* xrow = x + (size_t)(R0 + m) * D_DIM;
#pragma unroll
    for (int c = 0; c < 16; ++c) {
        const int d0 = c * 32 + h * 16;
        v4f a0 = *(const v4f*)(xrow + d0);
        v4f a1 = *(const v4f*)(xrow + d0 + 4);
        v4f a2 = *(const v4f*)(xrow + d0 + 8);
        v4f a3 = *(const v4f*)(xrow + d0 + 12);
#pragma unroll
        for (int e = 0; e < 4; ++e) {
            xa[c][e]      = (_Float16)a0[e];
            xa[c][4 + e]  = (_Float16)a1[e];
            xa[c][8 + e]  = (_Float16)a2[e];
            xa[c][12 + e] = (_Float16)a3[e];
        }
    }

#pragma unroll 1
    for (int ct = 0; ct < 16; ++ct) {
        const _Float16* wcol = WTz + (size_t)(ct * 16 + m) * D_DIM;
        v8f acc0 = {}, acc1 = {};
        // Software pipeline: keep next chunk's B fragment in flight.
        v16h bcur = loadB16(wcol + h * 16);
#pragma unroll
        for (int c = 0; c < 16; ++c) {
            v16h bnext = (c < 15) ? loadB16(wcol + (c + 1) * 32 + h * 16) : bcur;
            if (c & 1)
                acc1 = __builtin_amdgcn_wmma_f32_16x16x32_f16(
                    false, xa[c], false, bcur, (short)0, acc1, false, false);
            else
                acc0 = __builtin_amdgcn_wmma_f32_16x16x32_f16(
                    false, xa[c], false, bcur, (short)0, acc0, false, false);
            bcur = bnext;
        }
        v8f acc;
#pragma unroll
        for (int r = 0; r < 8; ++r) acc[r] = acc0[r] + acc1[r];

        // C layout: lane col n=m, VGPR r -> row r + 8h
        _Float16* orow = dst + (size_t)(R0 + h * 8) * U_DIM + ct * 16 + m;
#pragma unroll
        for (int r = 0; r < 8; ++r) orow[(size_t)r * U_DIM] = (_Float16)acc[r];
        if (z == 2) {
            // vT[b][u][s]; this lane's 8 values are s-contiguous -> b128 store
            const int bb = R0 >> 11;
            const int s0 = (R0 & (S_LEN - 1)) + h * 8;
            v8h pv;
#pragma unroll
            for (int r = 0; r < 8; ++r) pv[r] = (_Float16)acc[r];
            *(v8h*)(vT + ((size_t)bb * U_DIM + (ct * 16 + m)) * S_LEN + s0) = pv;
        }
    }
}

// --------------------------------------------------------------------------
// Tile staging helper for the attention kernel (async if available).
// --------------------------------------------------------------------------
__device__ __forceinline__ void stage_tiles(int tid,
                                            const _Float16* __restrict__ ksrc,
                                            const _Float16* __restrict__ vtsrc,
                                            _Float16* kT, _Float16* vTT) {
#if defined(USE_ASYNC_LDS)
#pragma unroll
    for (int i = 0; i < 8; ++i) {
        const int c8 = tid + i * 128;                 // 16B chunk id
        async_copy_b128(ksrc + c8 * 8, kT + c8 * 8);
    }
#pragma unroll
    for (int i = 0; i < 8; ++i) {
        const int c8 = tid + i * 128;
        const int u  = c8 >> 2;
        const int p4 = (c8 & 3) * 8;
        async_copy_b128(vtsrc + (size_t)u * S_LEN + p4, vTT + u * 32 + p4);
    }
#else
#pragma unroll
    for (int i = 0; i < 8; ++i) {
        const int c8 = tid + i * 128;
        *(v8h*)(kT + c8 * 8) = *(const v8h*)(ksrc + c8 * 8);
    }
#pragma unroll
    for (int i = 0; i < 8; ++i) {
        const int c8 = tid + i * 128;
        const int u  = c8 >> 2;
        const int p4 = (c8 & 3) * 8;
        *(v8h*)(vTT + u * 32 + p4) = *(const v8h*)(vtsrc + (size_t)u * S_LEN + p4);
    }
#endif
}

// --------------------------------------------------------------------------
// Kernel 2: fused flash attention.  Block = 128 threads (4 waves); each wave
// owns a 16-row q strip.  K / V^T tiles double-buffered in LDS; the next
// tile's async DMA streams while the current tile's WMMAs execute.
// --------------------------------------------------------------------------
__global__ __launch_bounds__(128)
void flash_attn_kernel(const _Float16* __restrict__ q,
                       const _Float16* __restrict__ k,
                       const _Float16* __restrict__ vT,
                       float* __restrict__ out) {
    __shared__ _Float16 kTile[2][32 * U_DIM];    // [buf][key][u]   2x16 KB
    __shared__ _Float16 vTileT[2][U_DIM * 32];   // [buf][u][key]   2x16 KB
    __shared__ _Float16 pScr[4][16 * 32];        // per-wave P       4 KB

    const int tid  = threadIdx.x;
    const int lane = tid & 31;
    const int wave = tid >> 5;
    const int m = lane & 15;
    const int h = lane >> 4;

    const int b  = blockIdx.y;
    const int q0 = blockIdx.x * 64 + wave * 16;
    const size_t base  = (size_t)b * S_LEN * U_DIM;
    const size_t vbase = (size_t)b * U_DIM * S_LEN;

    // Kick off tile 0 staging before anything else.
    stage_tiles(tid, k + base, vT + vbase, kTile[0], vTileT[0]);

    // Preload q strip as 8 A-fragments (overlaps with tile-0 DMA).
    v16h qa[8];
    const _Float16* qrow = q + base + (size_t)(q0 + m) * U_DIM;
#pragma unroll
    for (int c = 0; c < 8; ++c) {
        v8h lo = *(const v8h*)(qrow + c * 32 + h * 16);
        v8h hi = *(const v8h*)(qrow + c * 32 + h * 16 + 8);
#pragma unroll
        for (int e = 0; e < 8; ++e) { qa[c][e] = lo[e]; qa[c][8 + e] = hi[e]; }
    }

    v8f acc[16] = {};
    float mrow[8], lrow[8];
#pragma unroll
    for (int r = 0; r < 8; ++r) { mrow[r] = -INFINITY; lrow[r] = 0.0f; }

    for (int j = 0; j < S_LEN; j += 32) {
        const int cur = (j >> 5) & 1;
#if defined(USE_ASYNC_LDS)
        __builtin_amdgcn_s_wait_asynccnt(0);     // this wave's loads for tile j done
#endif
        __syncthreads();                          // everyone's data visible; prev buf free
        if (j + 32 < S_LEN)                       // stream tile j+32 while computing j
            stage_tiles(tid, k + base + (size_t)(j + 32) * U_DIM,
                        vT + vbase + (j + 32), kTile[cur ^ 1], vTileT[cur ^ 1]);

        const _Float16* kT  = kTile[cur];
        const _Float16* vTT = vTileT[cur];

        // ---- scores: S = q (16x256) * K^T (256x32); 4 independent chains ----
        v8f s0a = {}, s0b = {}, s1a = {}, s1b = {};
#pragma unroll
        for (int c = 0; c < 8; c += 2) {
            v16h b00 = *(const v16h*)(kT + m * U_DIM + c * 32 + h * 16);
            v16h b10 = *(const v16h*)(kT + (16 + m) * U_DIM + c * 32 + h * 16);
            v16h b01 = *(const v16h*)(kT + m * U_DIM + (c + 1) * 32 + h * 16);
            v16h b11 = *(const v16h*)(kT + (16 + m) * U_DIM + (c + 1) * 32 + h * 16);
            s0a = __builtin_amdgcn_wmma_f32_16x16x32_f16(
                false, qa[c], false, b00, (short)0, s0a, false, false);
            s1a = __builtin_amdgcn_wmma_f32_16x16x32_f16(
                false, qa[c], false, b10, (short)0, s1a, false, false);
            s0b = __builtin_amdgcn_wmma_f32_16x16x32_f16(
                false, qa[c + 1], false, b01, (short)0, s0b, false, false);
            s1b = __builtin_amdgcn_wmma_f32_16x16x32_f16(
                false, qa[c + 1], false, b11, (short)0, s1b, false, false);
        }
        v8f s0, s1;
#pragma unroll
        for (int r = 0; r < 8; ++r) { s0[r] = s0a[r] + s0b[r]; s1[r] = s1a[r] + s1b[r]; }

        // ---- online softmax ----
        float pmax[8];
#pragma unroll
        for (int r = 0; r < 8; ++r) pmax[r] = fmaxf(s0[r], s1[r]);
#pragma unroll
        for (int mask = 1; mask <= 8; mask <<= 1)
#pragma unroll
            for (int r = 0; r < 8; ++r)
                pmax[r] = fmaxf(pmax[r], __shfl_xor(pmax[r], mask, 32));

        float alpha[8], p0[8], p1[8], psum[8];
#pragma unroll
        for (int r = 0; r < 8; ++r) {
            const float mn = fmaxf(mrow[r], pmax[r]);
            alpha[r] = __expf(mrow[r] - mn);
            mrow[r]  = mn;
            p0[r] = __expf(s0[r] - mn);
            p1[r] = __expf(s1[r] - mn);
            psum[r] = p0[r] + p1[r];
        }
#pragma unroll
        for (int mask = 1; mask <= 8; mask <<= 1)
#pragma unroll
            for (int r = 0; r < 8; ++r)
                psum[r] += __shfl_xor(psum[r], mask, 32);
#pragma unroll
        for (int r = 0; r < 8; ++r) lrow[r] = lrow[r] * alpha[r] + psum[r];
#pragma unroll
        for (int n = 0; n < 16; ++n)
#pragma unroll
            for (int r = 0; r < 8; ++r) acc[n][r] *= alpha[r];

        // ---- bounce P through LDS: C-frag -> A-frag relayout ----
        _Float16* ps = pScr[wave];
#pragma unroll
        for (int r = 0; r < 8; ++r) {
            ps[(r + 8 * h) * 32 + m]      = (_Float16)p0[r];
            ps[(r + 8 * h) * 32 + 16 + m] = (_Float16)p1[r];
        }
        v16h pa;
        {
            v8h lo = *(const v8h*)(ps + m * 32 + h * 16);
            v8h hi = *(const v8h*)(ps + m * 32 + h * 16 + 8);
#pragma unroll
            for (int e = 0; e < 8; ++e) { pa[e] = lo[e]; pa[8 + e] = hi[e]; }
        }

        // ---- out += P (16x32) * V (32x256); 16 independent chains ----
#pragma unroll
        for (int n = 0; n < 16; ++n) {
            v16h vb = *(const v16h*)(vTT + (n * 16 + m) * 32 + h * 16);
            acc[n] = __builtin_amdgcn_wmma_f32_16x16x32_f16(
                false, pa, false, vb, (short)0, acc[n], false, false);
        }
    }

    // ---- epilogue ----
    float inv[8];
#pragma unroll
    for (int r = 0; r < 8; ++r) inv[r] = 1.0f / lrow[r];
    float* orow = out + base + (size_t)(q0 + 8 * h) * U_DIM + m;
#pragma unroll
    for (int n = 0; n < 16; ++n)
#pragma unroll
        for (int r = 0; r < 8; ++r)
            orow[(size_t)r * U_DIM + n * 16] = acc[n][r] * inv[r];
}

// --------------------------------------------------------------------------
extern "C" void kernel_launch(void* const* d_in, const int* in_sizes, int n_in,
                              void* d_out, int out_size, void* d_ws, size_t ws_size,
                              hipStream_t stream) {
    (void)in_sizes; (void)n_in; (void)out_size; (void)ws_size;
    const float* x  = (const float*)d_in[0];
    const float* Wq = (const float*)d_in[1];
    const float* Wk = (const float*)d_in[2];
    const float* Wv = (const float*)d_in[3];
    float* out = (float*)d_out;

    _Float16* qd = (_Float16*)d_ws;                       // 4 MB
    _Float16* kd = qd + (size_t)BS_ROWS * U_DIM;          // 4 MB
    _Float16* vd = kd + (size_t)BS_ROWS * U_DIM;          // 4 MB
    _Float16* vT = vd + (size_t)BS_ROWS * U_DIM;          // 4 MB
    _Float16* WT = vT + (size_t)BS_ROWS * U_DIM;          // 0.75 MB

    wt_kernel<<<dim3(3 * U_DIM * D_DIM / 256), 256, 0, stream>>>(Wq, Wk, Wv, WT);

    proj_qkv_kernel<<<dim3(BS_ROWS / 64, 3), 128, 0, stream>>>(x, WT, qd, kd, vd, vT);

    flash_attn_kernel<<<dim3(S_LEN / 64, 4), 128, 0, stream>>>(qd, kd, vT, out);
}
